// MultiHeadShogiAttention_58815282152337
// MI455X (gfx1250) — matrix operand order repaired and checked by reading, not compiled, not run on hardware
//
#include <hip/hip_runtime.h>

// ---------------------------------------------------------------------------
// MultiHeadShogiAttention for MI455X (gfx1250, wave32, WMMA)
//
//   DIM=512, HEADS=8, HD=64, N=81 (pad->96), B=1024, M = B*N = 82944
//   Stage 1: qkvh[M][1536]  = f16( x[M][512] @ qkv_w + qkv_b )       (WMMA f16)
//   Stage 2: ctx[M][512]    = fused attention per (b,h) in LDS       (WMMA f16)
//   Stage 3: out[M][512]    = ctx @ out_w + out_b                    (WMMA f16)
// ---------------------------------------------------------------------------

typedef __attribute__((ext_vector_type(16))) _Float16 v16h;
typedef __attribute__((ext_vector_type(4)))  _Float16 v4h;
typedef __attribute__((ext_vector_type(8)))  float    v8f;

#define DIMK   512
#define HEADS  8
#define HD     64
#define NTOK   81
#define NPAD   96
#define BATCH  1024
#define MROWS  (BATCH * NTOK)   // 82944 (divisible by 128)
#define QKVN   (3 * DIMK)       // 1536
#define NEGF   (-1.0e9f)
#define NEGPAD (-3.0e9f)

__device__ __forceinline__ v16h load_frag16(const _Float16* p0, const _Float16* p1) {
    v16h r;
    ((uint4*)&r)[0] = *(const uint4*)p0;
    ((uint4*)&r)[1] = *(const uint4*)p1;
    return r;
}

// ---------------------------------------------------------------------------
// Kernel 0: convert + transpose weights to f16 [n][k] (B-fragment layout
// wants contiguous K per output column n).
// ---------------------------------------------------------------------------
__global__ void shogi_convert_weights(const float* __restrict__ qkv_w,
                                      const float* __restrict__ out_w,
                                      _Float16* __restrict__ qkv_wt,
                                      _Float16* __restrict__ out_wt) {
    int i = blockIdx.x * 256 + threadIdx.x;          // grid covers exactly 1048576
    if (i < DIMK * QKVN) {                           // 786432 dest elems [n][k]
        int n = i >> 9, k = i & 511;
        qkv_wt[i] = (_Float16)qkv_w[(size_t)k * QKVN + n];
    } else {
        int j = i - DIMK * QKVN;                     // 262144 dest elems [n][k]
        int n = j >> 9, k = j & 511;
        out_wt[j] = (_Float16)out_w[(size_t)k * DIMK + n];
    }
}

// ---------------------------------------------------------------------------
// Kernel 1: QKV projection. C[M][1536] f16 = f32 x[M][512] @ wt^T + b.
// 256 threads = 8 waves (4x2), block tile 128x128, K-step 32.
// ---------------------------------------------------------------------------
__global__ __launch_bounds__(256) void shogi_qkv_gemm(
        const float* __restrict__ x, const _Float16* __restrict__ wt,
        const float* __restrict__ bias, _Float16* __restrict__ yh) {
    __shared__ _Float16 As[128 * 32];   // [m][k] row-major
    __shared__ _Float16 Bs[128 * 32];   // [n][k] row-major (pre-transposed W)

    const int m0   = blockIdx.x * 128;
    const int n0   = blockIdx.y * 128;
    const int tid  = threadIdx.x;
    const int lane = tid & 31, wv = tid >> 5;
    const int wm   = wv >> 1,  wn = wv & 1;     // 4x2 wave grid
    const int hi   = lane >> 4, ln = lane & 15;

    v8f acc[2][4] = {};

    for (int kk = 0; kk < DIMK; kk += 32) {
        // stage A tile: 128x32 f32 -> f16
#pragma unroll
        for (int i = 0; i < 4; ++i) {
            int idx = tid + i * 256;                 // 0..1023
            int row = idx >> 3, c = (idx & 7) << 2;  // 8 float4 per row
            float4 f = *(const float4*)(x + (size_t)(m0 + row) * DIMK + kk + c);
            v4h h4 = { (_Float16)f.x, (_Float16)f.y, (_Float16)f.z, (_Float16)f.w };
            *(v4h*)&As[row * 32 + c] = h4;
        }
        // stage B tile: 128 cols x 32 k, f16, contiguous
#pragma unroll
        for (int i = 0; i < 2; ++i) {
            int idx = tid + i * 256;                 // 0..511
            int row = idx >> 2, c = (idx & 3) << 3;  // 4 uint4 per row
            *(uint4*)&Bs[row * 32 + c] =
                *(const uint4*)(wt + (size_t)(n0 + row) * DIMK + kk + c);
        }
        __syncthreads();

        v16h a[2];
#pragma unroll
        for (int mt = 0; mt < 2; ++mt) {
            // A frag (16x32 f16): lane row = ln; K chunks {hi*8, hi*8+16}
            const _Float16* p = &As[(wm * 32 + mt * 16 + ln) * 32 + hi * 8];
            a[mt] = load_frag16(p, p + 16);
        }
#pragma unroll
        for (int nt = 0; nt < 4; ++nt) {
            // B frag (32x16 f16): lane col = ln; K chunks {hi*8, hi*8+16}
            const _Float16* p = &Bs[(wn * 64 + nt * 16 + ln) * 32 + hi * 8];
            v16h bf = load_frag16(p, p + 16);
#pragma unroll
            for (int mt = 0; mt < 2; ++mt)
                acc[mt][nt] = __builtin_amdgcn_wmma_f32_16x16x32_f16(
                    false, a[mt], false, bf, (short)0, acc[mt][nt], false, false);
        }
        __syncthreads();
    }

    // epilogue: +bias, convert to f16. C layout: lane hi-half holds rows +8.
#pragma unroll
    for (int nt = 0; nt < 4; ++nt) {
        int col = n0 + wn * 64 + nt * 16 + ln;
        float bv = bias[col];
#pragma unroll
        for (int mt = 0; mt < 2; ++mt) {
            int rbase = m0 + wm * 32 + mt * 16 + hi * 8;
#pragma unroll
            for (int r = 0; r < 8; ++r)
                yh[(size_t)(rbase + r) * QKVN + col] = (_Float16)(acc[mt][nt][r] + bv);
        }
    }
}

// ---------------------------------------------------------------------------
// Kernel 2: fused attention per (b,h). 192 threads = 6 waves; wave w owns
// q-rows [16w, 16w+16). Keys padded 81->96 (pad bias -3e9 -> exp == 0).
// ---------------------------------------------------------------------------
__global__ __launch_bounds__(192) void shogi_attention(
        const _Float16* __restrict__ qkvh, const int* __restrict__ msk,
        const float* __restrict__ ray, _Float16* __restrict__ ctx) {
    __shared__ _Float16 Qs[NPAD * HD];      // [qrow][d]
    __shared__ _Float16 Ks[NPAD * HD];      // [key][d]
    __shared__ _Float16 Vts[HD * NPAD];     // [d][key]  (transposed V)
    __shared__ _Float16 Ps[6 * 16 * NPAD];  // per-wave probs [16][96]
    __shared__ float    smaskf[NPAD];

    const int bh = blockIdx.x;
    const int b  = bh >> 3, h = bh & 7;
    const int tid  = threadIdx.x;
    const int lane = tid & 31, wv = tid >> 5;
    const int hi   = lane >> 4, ln = lane & 15;
    const int w16  = wv * 16;

    const _Float16* qp = qkvh + (size_t)(b * NTOK) * QKVN + h * HD;          // Q cols
    const _Float16* kp = qp + DIMK;                                          // K cols
    const _Float16* vp = qp + 2 * DIMK;                                      // V cols

    // ---- stage Q/K row-major, V transposed (zero pad rows/keys 81..95) ----
#pragma unroll
    for (int i = 0; i < 8; ++i) {
        int idx = tid + i * 192;                 // 0..1535
        int row = idx >> 4, c = (idx & 15) << 2; // 4 halfs along d
        uint2 zq = {0u, 0u}, zk = {0u, 0u};
        if (row < NTOK) {
            zq = *(const uint2*)(qp + (size_t)row * QKVN + c);
            zk = *(const uint2*)(kp + (size_t)row * QKVN + c);
        }
        *(uint2*)&Qs[row * HD + c] = zq;
        *(uint2*)&Ks[row * HD + c] = zk;
        if (row < NTOK) {
            const _Float16* vs = vp + (size_t)row * QKVN + c;
            Vts[(c + 0) * NPAD + row] = vs[0];
            Vts[(c + 1) * NPAD + row] = vs[1];
            Vts[(c + 2) * NPAD + row] = vs[2];
            Vts[(c + 3) * NPAD + row] = vs[3];
        } else {
            Vts[(c + 0) * NPAD + row] = (_Float16)0.f;
            Vts[(c + 1) * NPAD + row] = (_Float16)0.f;
            Vts[(c + 2) * NPAD + row] = (_Float16)0.f;
            Vts[(c + 3) * NPAD + row] = (_Float16)0.f;
        }
    }
    if (tid < NPAD)
        smaskf[tid] = (tid < NTOK) ? (msk[b * NTOK + tid] ? 0.f : NEGF) : NEGPAD;
    __syncthreads();

    // ---- S = Q K^T : 6 tiles of 16x16, head-dim in 2 chunks of 32 ----
    v16h aq[2];
#pragma unroll
    for (int kc = 0; kc < 2; ++kc) {
        const _Float16* p = &Qs[(w16 + ln) * HD + kc * 32 + hi * 8];
        aq[kc] = load_frag16(p, p + 16);
    }
    v8f s[6];
#pragma unroll
    for (int t = 0; t < 6; ++t) {
        v8f c = {};
#pragma unroll
        for (int kc = 0; kc < 2; ++kc) {
            const _Float16* p = &Ks[(t * 16 + ln) * HD + kc * 32 + hi * 8];
            v16h bk = load_frag16(p, p + 16);
            c = __builtin_amdgcn_wmma_f32_16x16x32_f16(
                    false, aq[kc], false, bk, (short)0, c, false, false);
        }
        s[t] = c;
    }

    // ---- scale + ray bias + key-padding mask, softmax, P -> LDS f16 ----
    const float scale = 0.125f;                    // HD^-0.5
#pragma unroll
    for (int r = 0; r < 8; ++r) {
        int m = r + hi * 8;
        int qrow = w16 + m;
        bool qok = qrow < NTOK;
        float rmax = -3.4e38f;
#pragma unroll
        for (int t = 0; t < 6; ++t) {
            int j = t * 16 + ln;
            float v;
            if (j < NTOK) {
                if (smaskf[j] != 0.f) {
                    v = NEGF;                       // exact reference semantics
                } else {
                    float rv = qok ? ray[qrow * NTOK + j] : 0.f;
                    v = s[t][r] * scale + rv;
                }
            } else {
                v = NEGPAD;                         // pad keys vanish after max-sub
            }
            s[t][r] = v;
            rmax = fmaxf(rmax, v);
        }
        // reduce within 16-lane half (the half owns one row of the tile)
        for (int off = 1; off < 16; off <<= 1)
            rmax = fmaxf(rmax, __shfl_xor(rmax, off, 32));
        float sum = 0.f;
#pragma unroll
        for (int t = 0; t < 6; ++t) {
            float p = __expf(s[t][r] - rmax);
            s[t][r] = p;
            sum += p;
        }
        for (int off = 1; off < 16; off <<= 1)
            sum += __shfl_xor(sum, off, 32);
        float inv = 1.f / sum;
        _Float16* pr = &Ps[wv * (16 * NPAD) + m * NPAD + ln];
#pragma unroll
        for (int t = 0; t < 6; ++t)
            pr[t * 16] = (_Float16)(s[t][r] * inv);
    }
    // wave reads back only its own P region: same-wave LDS ordering suffices.

    // ---- O = P V : 4 d-tiles, keys in 3 chunks of 32 ----
    v16h ap[3];
#pragma unroll
    for (int kc = 0; kc < 3; ++kc) {
        const _Float16* p = &Ps[wv * (16 * NPAD) + ln * NPAD + kc * 32 + hi * 8];
        ap[kc] = load_frag16(p, p + 16);
    }
#pragma unroll
    for (int dt = 0; dt < 4; ++dt) {
        v8f o = {};
#pragma unroll
        for (int kc = 0; kc < 3; ++kc) {
            const _Float16* p = &Vts[(dt * 16 + ln) * NPAD + kc * 32 + hi * 8];
            v16h bv = load_frag16(p, p + 16);
            o = __builtin_amdgcn_wmma_f32_16x16x32_f16(
                    false, ap[kc], false, bv, (short)0, o, false, false);
        }
#pragma unroll
        for (int r = 0; r < 8; ++r) {
            int qrow = w16 + r + hi * 8;
            if (qrow < NTOK)
                ctx[(size_t)(b * NTOK + qrow) * DIMK + h * HD + dt * 16 + ln] =
                    (_Float16)o[r];
        }
    }
}

// ---------------------------------------------------------------------------
// Kernel 3: output projection. out[M][512] f32 = ctx[M][512] f16 @ wt^T + b.
// ---------------------------------------------------------------------------
__global__ __launch_bounds__(256) void shogi_out_gemm(
        const _Float16* __restrict__ ah, const _Float16* __restrict__ wt,
        const float* __restrict__ bias, float* __restrict__ y) {
    __shared__ _Float16 As[128 * 32];
    __shared__ _Float16 Bs[128 * 32];

    const int m0   = blockIdx.x * 128;
    const int n0   = blockIdx.y * 128;
    const int tid  = threadIdx.x;
    const int lane = tid & 31, wv = tid >> 5;
    const int wm   = wv >> 1,  wn = wv & 1;
    const int hi   = lane >> 4, ln = lane & 15;

    v8f acc[2][4] = {};

    for (int kk = 0; kk < DIMK; kk += 32) {
#pragma unroll
        for (int i = 0; i < 2; ++i) {
            int idx = tid + i * 256;
            int row = idx >> 2, c = (idx & 3) << 3;
            *(uint4*)&As[row * 32 + c] =
                *(const uint4*)(ah + (size_t)(m0 + row) * DIMK + kk + c);
        }
#pragma unroll
        for (int i = 0; i < 2; ++i) {
            int idx = tid + i * 256;
            int row = idx >> 2, c = (idx & 3) << 3;
            *(uint4*)&Bs[row * 32 + c] =
                *(const uint4*)(wt + (size_t)(n0 + row) * DIMK + kk + c);
        }
        __syncthreads();

        v16h a[2];
#pragma unroll
        for (int mt = 0; mt < 2; ++mt) {
            const _Float16* p = &As[(wm * 32 + mt * 16 + ln) * 32 + hi * 8];
            a[mt] = load_frag16(p, p + 16);
        }
#pragma unroll
        for (int nt = 0; nt < 4; ++nt) {
            const _Float16* p = &Bs[(wn * 64 + nt * 16 + ln) * 32 + hi * 8];
            v16h bf = load_frag16(p, p + 16);
#pragma unroll
            for (int mt = 0; mt < 2; ++mt)
                acc[mt][nt] = __builtin_amdgcn_wmma_f32_16x16x32_f16(
                    false, a[mt], false, bf, (short)0, acc[mt][nt], false, false);
        }
        __syncthreads();
    }

#pragma unroll
    for (int nt = 0; nt < 4; ++nt) {
        int col = n0 + wn * 64 + nt * 16 + ln;
        float bv = bias[col];
#pragma unroll
        for (int mt = 0; mt < 2; ++mt) {
            int rbase = m0 + wm * 32 + mt * 16 + hi * 8;
#pragma unroll
            for (int r = 0; r < 8; ++r)
                y[(size_t)(rbase + r) * DIMK + col] = acc[mt][nt][r] + bv;
        }
    }
}

// ---------------------------------------------------------------------------
extern "C" void kernel_launch(void* const* d_in, const int* in_sizes, int n_in,
                              void* d_out, int out_size, void* d_ws, size_t ws_size,
                              hipStream_t stream) {
    (void)in_sizes; (void)n_in; (void)out_size; (void)ws_size;

    const float* x     = (const float*)d_in[0];
    const int*   msk   = (const int*)d_in[1];
    const float* qkv_w = (const float*)d_in[2];
    const float* qkv_b = (const float*)d_in[3];
    const float* out_w = (const float*)d_in[4];
    const float* out_b = (const float*)d_in[5];
    const float* ray   = (const float*)d_in[6];
    float*       out   = (float*)d_out;

    // workspace carve (f16, all 16B-aligned): ~326 MB total
    _Float16* qkv_wt = (_Float16*)d_ws;                         // 512*1536
    _Float16* out_wt = qkv_wt + (size_t)DIMK * QKVN;            // 512*512
    _Float16* qkvh   = out_wt + (size_t)DIMK * DIMK;            // 82944*1536
    _Float16* ctx    = qkvh + (size_t)MROWS * QKVN;             // 82944*512

    shogi_convert_weights<<<4096, 256, 0, stream>>>(qkv_w, out_w, qkv_wt, out_wt);
    shogi_qkv_gemm<<<dim3(MROWS / 128, QKVN / 128), 256, 0, stream>>>(
        x, qkv_wt, qkv_b, qkvh);
    shogi_attention<<<BATCH * HEADS, 192, 0, stream>>>(qkvh, msk, ray, ctx);
    shogi_out_gemm<<<dim3(MROWS / 128, DIMK / 128), 256, 0, stream>>>(
        ctx, out_wt, out_b, out);
}
